// GCNEncoder_30571577213152
// MI455X (gfx1250) — matrix operand order repaired and checked by reading, not compiled, not run on hardware
//
#include <hip/hip_runtime.h>
#include <hip/hip_bf16.h>
#include <stdint.h>

// ---------------------------------------------------------------------------
// GCN 2-layer encoder for MI455X (gfx1250, wave32).
//   layer: S = H @ W (bf16 WMMA, f32 accum)
//          AGG[dst] += S[src] * ew       (edge-parallel native f32 atomics)
//          H' = leaky_relu(AGG + b)
// ---------------------------------------------------------------------------

typedef __attribute__((ext_vector_type(16))) __bf16 v16bf;
typedef __attribute__((ext_vector_type(8)))  float  v8f;

#define IN_DIM 256
#define HID1   256
#define HID2   128
#define NEG_SLOPE 0.01f
#define EPW    4          // edges per wave in the scatter kernel

static __device__ __forceinline__ unsigned short f2bf(float f) {
  unsigned int u = __float_as_uint(f);
  u += 0x7fffu + ((u >> 16) & 1u);   // round-to-nearest-even
  return (unsigned short)(u >> 16);
}

// Native no-return f32 atomic add at device scope (L2 atomic units).
static __device__ __forceinline__ void atomic_add_f32_dev(float* p, float v) {
  asm volatile("global_atomic_add_f32 %0, %1, off scope:SCOPE_DEV"
               :: "v"(p), "v"(v) : "memory");
}

// ---- elementwise f32 -> bf16 ----------------------------------------------
__global__ void k_f32_to_bf16(const float* __restrict__ in,
                              unsigned short* __restrict__ out, int n) {
  int i = blockIdx.x * blockDim.x + threadIdx.x;
  if (i < n) out[i] = f2bf(in[i]);
}

// ---- weight transpose+convert: Wt[n*K + k] = bf16(W[k*N + n]) --------------
__global__ void k_transpose_bf16(const float* __restrict__ W,
                                 unsigned short* __restrict__ Wt,
                                 int K, int N) {
  int i = blockIdx.x * blockDim.x + threadIdx.x;
  if (i >= K * N) return;
  int k = i / N, n = i % N;
  Wt[n * K + k] = f2bf(W[i]);
}

// ---- bf16 GEMM: C[M,N] = A[M,K] * Bt[N,K]^T  (wave = 16x64 strip) ----------
// A fragment (16-bit 16x32, ISA 7.12.2): lane m = lane&15, half = lane>>4,
//   elems = A[m][kb + 8*half .. +8) ++ A[m][kb + 16 + 8*half .. +8)
// B fragment: lane n = lane&15, elems = Bt[n][kb + 8*half .. +8) ++ Bt[n][kb + 16 + 8*half .. +8)
// C fragment: c[r] -> C[mt*16 + 8*half + r][n]
__global__ void __launch_bounds__(256)
k_gemm_bf16(const unsigned short* __restrict__ A,
            const unsigned short* __restrict__ Bt,
            float* __restrict__ C, int M, int N, int K) {
  const int wave = threadIdx.x >> 5;
  const int lane = threadIdx.x & 31;
  const int half = lane >> 4;
  const int l16  = lane & 15;
  const int mt   = blockIdx.x * 8 + wave;      // wave-uniform
  if (mt * 16 >= M) return;                    // uniform exit: EXEC all-ones kept
  const int n0 = blockIdx.y * 64;

  const unsigned short* arow = A + (size_t)(mt * 16 + l16) * K;

  v8f acc0 = 0.f, acc1 = 0.f, acc2 = 0.f, acc3 = 0.f;

  for (int kb = 0; kb < K; kb += 32) {
    union { uint4 u[2]; v16bf v; } a;
    a.u[0] = *(const uint4*)(arow + kb + 8 * half);
    a.u[1] = *(const uint4*)(arow + kb + 16 + 8 * half);

#pragma unroll
    for (int t = 0; t < 4; ++t) {
      const unsigned short* brow = Bt + (size_t)(n0 + t * 16 + l16) * K;
      union { uint4 u[2]; v16bf v; } b;
      b.u[0] = *(const uint4*)(brow + kb + 8 * half);
      b.u[1] = *(const uint4*)(brow + kb + 16 + 8 * half);
      v8f c = (t == 0) ? acc0 : (t == 1) ? acc1 : (t == 2) ? acc2 : acc3;
      c = __builtin_amdgcn_wmma_f32_16x16x32_bf16(
              false, a.v, false, b.v, (short)0, c, false, false);
      if (t == 0) acc0 = c; else if (t == 1) acc1 = c;
      else if (t == 2) acc2 = c; else acc3 = c;
    }
  }

#pragma unroll
  for (int t = 0; t < 4; ++t) {
    v8f c = (t == 0) ? acc0 : (t == 1) ? acc1 : (t == 2) ? acc2 : acc3;
    float* crow = C + (size_t)(mt * 16 + 8 * half) * N + (n0 + t * 16 + l16);
#pragma unroll
    for (int r = 0; r < 8; ++r) crow[(size_t)r * N] = c[r];
  }
}

// ---- edge-parallel SpMM scatter: agg[dst[e]] += S[src[e]] * ew[e] ----------
// One wave handles EPW edges. All source rows are prefetched first
// (global_prefetch_b8; lanes stride D/32 floats so each 128B line is touched), then
// lanes stream coalesced float4 chunks and issue native f32 atomics.
__global__ void k_spmm_scatter(const float* __restrict__ S,
                               const int* __restrict__ src,
                               const int* __restrict__ dst,
                               const float* __restrict__ ew,
                               float* __restrict__ agg, int E, int D) {
  int gw   = (int)(((size_t)blockIdx.x * blockDim.x + threadIdx.x) >> 5);
  int lane = threadIdx.x & 31;
  int e0   = gw * EPW;
  if (e0 >= E) return;                          // wave-uniform
  int elim = (E - e0 < EPW) ? (E - e0) : EPW;   // wave-uniform

  int stride = D >> 5;                          // floats per lane for prefetch
#pragma unroll
  for (int e = 0; e < EPW; ++e) {
    if (e < elim) {
      const float* srow = S + (size_t)src[e0 + e] * D;
      __builtin_prefetch(srow + lane * stride, 0, 3);
    }
  }

  for (int e = 0; e < elim; ++e) {
    int   s = src[e0 + e];
    int   d = dst[e0 + e];
    float w = ew[e0 + e];
    const float* srow = S + (size_t)s * D;
    float* drow = agg + (size_t)d * D;
    for (int c = lane * 4; c < D; c += 128) {
      float4 v = *(const float4*)(srow + c);
      atomic_add_f32_dev(drow + c + 0, v.x * w);
      atomic_add_f32_dev(drow + c + 1, v.y * w);
      atomic_add_f32_dev(drow + c + 2, v.z * w);
      atomic_add_f32_dev(drow + c + 3, v.w * w);
    }
  }
}

// ---- bias + leaky_relu, emit bf16 for next layer's GEMM --------------------
__global__ void k_bias_leaky_bf16(const float* __restrict__ agg,
                                  const float* __restrict__ bias,
                                  unsigned short* __restrict__ out,
                                  int total, int D) {
  int i = blockIdx.x * blockDim.x + threadIdx.x;
  if (i >= total) return;
  float v = agg[i] + bias[i & (D - 1)];
  v = (v > 0.f) ? v : NEG_SLOPE * v;
  out[i] = f2bf(v);
}

// ---- bias + leaky_relu in-place f32 (final output) -------------------------
__global__ void k_bias_leaky_f32(float* __restrict__ out,
                                 const float* __restrict__ bias,
                                 int total, int D) {
  int i = blockIdx.x * blockDim.x + threadIdx.x;
  if (i >= total) return;
  float v = out[i] + bias[i & (D - 1)];
  out[i] = (v > 0.f) ? v : NEG_SLOPE * v;
}

// ---------------------------------------------------------------------------
extern "C" void kernel_launch(void* const* d_in, const int* in_sizes, int n_in,
                              void* d_out, int out_size, void* d_ws, size_t ws_size,
                              hipStream_t stream) {
  const float* x   = (const float*)d_in[0];
  const float* w1  = (const float*)d_in[1];
  const float* b1  = (const float*)d_in[2];
  const float* w2  = (const float*)d_in[3];
  const float* b2  = (const float*)d_in[4];
  const int*   src = (const int*)d_in[5];
  const int*   dst = (const int*)d_in[6];
  const float* ew  = (const float*)d_in[7];

  const int N = in_sizes[0] / IN_DIM;   // 50000 (multiple of 16)
  const int E = in_sizes[5];            // 800000
  float* out = (float*)d_out;

  // workspace carve-out (aligned to 256B)
  char*  ws  = (char*)d_ws;
  size_t off = 0;
  auto carve = [&](size_t bytes) -> void* {
    void* p = ws + off;
    off = (off + bytes + 255) & ~(size_t)255;
    return p;
  };
  unsigned short* xbf  = (unsigned short*)carve((size_t)N * IN_DIM * 2); // reused as h1bf
  unsigned short* wt1  = (unsigned short*)carve((size_t)IN_DIM * HID1 * 2);
  unsigned short* wt2  = (unsigned short*)carve((size_t)HID1 * HID2 * 2);
  float*          s1   = (float*)carve((size_t)N * HID1 * 4);            // reused as s2
  float*          agg1 = (float*)carve((size_t)N * HID1 * 4);
  unsigned short* h1bf = xbf;   // x no longer needed after GEMM1
  float*          s2   = s1;    // s1 no longer needed after scatter1

  const int TPB = 256;

  // --- precision conversion / weight transpose
  k_f32_to_bf16<<<(N * IN_DIM + TPB - 1) / TPB, TPB, 0, stream>>>(x, xbf, N * IN_DIM);
  k_transpose_bf16<<<(IN_DIM * HID1 + TPB - 1) / TPB, TPB, 0, stream>>>(w1, wt1, IN_DIM, HID1);
  k_transpose_bf16<<<(HID1 * HID2 + TPB - 1) / TPB, TPB, 0, stream>>>(w2, wt2, HID1, HID2);

  const int mblocks = ((N / 16) + 7) / 8;   // 8 waves (M-tiles) per block
  const int ewaves  = (E + EPW - 1) / EPW;
  const int eblocks = (ewaves * 32 + TPB - 1) / TPB;

  // --- layer 1
  k_gemm_bf16<<<dim3(mblocks, HID1 / 64), TPB, 0, stream>>>(xbf, wt1, s1, N, HID1, IN_DIM);
  hipMemsetAsync(agg1, 0, (size_t)N * HID1 * 4, stream);
  k_spmm_scatter<<<eblocks, TPB, 0, stream>>>(s1, src, dst, ew, agg1, E, HID1);
  k_bias_leaky_bf16<<<(N * HID1 + TPB - 1) / TPB, TPB, 0, stream>>>(agg1, b1, h1bf, N * HID1, HID1);

  // --- layer 2 (aggregate straight into d_out)
  k_gemm_bf16<<<dim3(mblocks, HID2 / 64), TPB, 0, stream>>>(h1bf, wt2, s2, N, HID2, HID1);
  hipMemsetAsync(out, 0, (size_t)N * HID2 * 4, stream);
  k_spmm_scatter<<<eblocks, TPB, 0, stream>>>(s2, src, dst, ew, out, E, HID2);
  k_bias_leaky_f32<<<(N * HID2 + TPB - 1) / TPB, TPB, 0, stream>>>(out, b2, N * HID2, HID2);
}
